// MoE_87067577024918
// MI455X (gfx1250) — matrix operand order repaired and checked by reading, not compiled, not run on hardware
//
#include <hip/hip_runtime.h>
#include <hip/hip_bf16.h>

#define TOK_N 4096          // B*S tokens
#define DIM_D 512
#define DIM_H 1024
#define NUM_E 8
#define NK_TOTAL (TOK_N * 2)   // total routed (token,expert) rows, exact for top-2

typedef __attribute__((ext_vector_type(16))) __bf16 v16bf;
typedef __attribute__((ext_vector_type(8)))  float  v8f;
typedef unsigned int v4u __attribute__((ext_vector_type(4)));
typedef int v4i_a __attribute__((vector_size(16)));   // matches async-LDS builtin param

union FragU { v4u q[2]; v16bf v; };

#if defined(__has_builtin)
#if __has_builtin(__builtin_amdgcn_global_load_async_to_lds_b128)
#define ASYNC_LDS 1
#endif
#endif

// Stage 16 bytes global -> LDS (async, tracked by ASYNCcnt).
__device__ __forceinline__ void stage16(const __bf16* gsrc, char* lds_dst) {
#ifdef ASYNC_LDS
  __builtin_amdgcn_global_load_async_to_lds_b128(
      (__attribute__((address_space(1))) v4i_a*)(void*)gsrc,
      (__attribute__((address_space(3))) v4i_a*)(void*)lds_dst,
      0, 0);
#else
  *(v4u*)lds_dst = *(const v4u*)gsrc;
#endif
}

// Drain async copies (ASYNCcnt -> 0) then workgroup barrier.
__device__ __forceinline__ void stage_fence() {
#ifdef ASYNC_LDS
#if __has_builtin(__builtin_amdgcn_s_wait_asynccnt)
  __builtin_amdgcn_s_wait_asynccnt(0);
#else
  asm volatile("s_wait_asynccnt 0x0" ::: "memory");
#endif
#endif
  __syncthreads();
}

// A fragment (16x32 bf16, M x K): lane l -> g = l>>4, m = l&15.
// halfs 0..7 : k = kb + g*8 + j ; halfs 8..15 : k = kb + 16 + g*8 + (j-8)
__device__ __forceinline__ v16bf load_a_frag(const __bf16* __restrict__ rowptr, int kb, int g) {
  FragU f;
  f.q[0] = *(const v4u*)(rowptr + kb + g * 8);
  f.q[1] = *(const v4u*)(rowptr + kb + 16 + g * 8);
  return f.v;
}

// B fragment from an LDS tile laid out as [n (64 rows)][32 k halfs] (64B per n-row).
// lane l -> g = l>>4, n = l&15 ; halfs 0..7 : k = g*8 + j ; halfs 8..15 : k = 16 + g*8 + (j-8)
__device__ __forceinline__ v16bf load_b_lds(const char* __restrict__ tile, int nf, int q, int g) {
  FragU f;
  const char* p = tile + (size_t)(nf + q) * 64 + g * 16;
  f.q[0] = *(const v4u*)(p);
  f.q[1] = *(const v4u*)(p + 32);
  return f.v;
}

__device__ __forceinline__ v8f wmma_bf16(v16bf a, v16bf b, v8f c) {
  return __builtin_amdgcn_wmma_f32_16x16x32_bf16(
      false, a, false, b, (short)0, c, false, false);
}

// Fast SiLU: single v_exp_f32 + v_rcp_f32 (no IEEE divide expansion).
__device__ __forceinline__ float silu_f(float v) {
  return v * __builtin_amdgcn_rcpf(1.0f + __expf(-v));
}

// ---------------- f32 -> bf16 cast ----------------
__global__ __launch_bounds__(256) void k_cvt_bf16(const float* __restrict__ src,
                                                  __bf16* __restrict__ dst, int n) {
  int i = blockIdx.x * 256 + threadIdx.x;
  if (i < n) dst[i] = (__bf16)src[i];
}

__global__ void k_zero_i32(int* __restrict__ p, int n) {
  int i = blockIdx.x * blockDim.x + threadIdx.x;
  if (i < n) p[i] = 0;
}

// ---------------- Router: one wave32 per token ----------------
__global__ __launch_bounds__(256) void k_router(
    const float* __restrict__ x, const float* __restrict__ gw,
    int* __restrict__ idx0, int* __restrict__ idx1,
    float* __restrict__ w0t, float* __restrict__ w1t,
    int* __restrict__ counts) {
  int wid  = (blockIdx.x * 256 + threadIdx.x) >> 5;
  int lane = threadIdx.x & 31;
  if (wid >= TOK_N) return;
  const float* xp = x + (size_t)wid * DIM_D;
  float acc[NUM_E];
#pragma unroll
  for (int e = 0; e < NUM_E; ++e) acc[e] = 0.f;
  for (int d = lane; d < DIM_D; d += 32) {
    float xv = xp[d];
#pragma unroll
    for (int e = 0; e < NUM_E; ++e)
      acc[e] = fmaf(xv, gw[e * DIM_D + d], acc[e]);
  }
#pragma unroll
  for (int e = 0; e < NUM_E; ++e) {
#pragma unroll
    for (int off = 16; off > 0; off >>= 1)
      acc[e] += __shfl_xor(acc[e], off, 32);
  }
  if (lane == 0) {
    int e0 = 0; float v0 = acc[0];
#pragma unroll
    for (int e = 1; e < NUM_E; ++e) if (acc[e] > v0) { v0 = acc[e]; e0 = e; }
    int e1 = (e0 == 0) ? 1 : 0; float v1 = acc[e1];
#pragma unroll
    for (int e = 0; e < NUM_E; ++e)
      if (e != e0 && acc[e] > v1) { v1 = acc[e]; e1 = e; }
    float p0 = __builtin_amdgcn_rcpf(1.f + __expf(v1 - v0)); // softmax over the 2 logits
    idx0[wid] = e0; idx1[wid] = e1;
    w0t[wid]  = p0; w1t[wid]  = 1.f - p0;
    atomicAdd(&counts[e0], 1);
    atomicAdd(&counts[e1], 1);
  }
}

__global__ void k_offsets(const int* __restrict__ counts, int* __restrict__ offs) {
  if (threadIdx.x == 0 && blockIdx.x == 0) {
    int s = 0;
    for (int e = 0; e < NUM_E; ++e) { offs[e] = s; s += counts[e]; }
  }
}

__global__ __launch_bounds__(256) void k_fill(
    const int* __restrict__ idx0, const int* __restrict__ idx1,
    int* __restrict__ fill, const int* __restrict__ offs,
    int* __restrict__ rows, int* __restrict__ slot0, int* __restrict__ slot1) {
  int t = blockIdx.x * 256 + threadIdx.x;
  if (t >= TOK_N) return;
  int e0 = idx0[t];
  int p  = atomicAdd(&fill[e0], 1);
  int r  = offs[e0] + p;
  rows[r] = t; slot0[t] = r;
  int e1 = idx1[t];
  p = atomicAdd(&fill[e1], 1);
  r = offs[e1] + p;
  rows[r] = t; slot1[t] = r;
}

// =====================================================================
// Grouped GEMM1: h = silu(X_gathered @ w1[e]^T) -> bf16
// block = 256 thr (8 waves); block tile = 256 M x 64 N; wave tile = 32 M x 64 N
// B tile (64 n x 32 k) async-staged in LDS, double buffered, last K-step peeled.
// grid.x = NUM_E * 16, grid.y = DIM_H/64
// =====================================================================
__global__ __launch_bounds__(256) void k_expert_gemm1(
    const __bf16* __restrict__ xbf, const __bf16* __restrict__ w1bf,
    const int* __restrict__ rows, const int* __restrict__ counts,
    const int* __restrict__ offs, __bf16* __restrict__ hbuf) {
  __shared__ __align__(32) char smem[2 * 4096];
  int e  = blockIdx.x >> 4;
  int mt = blockIdx.x & 15;
  int cnt = counts[e];
  int m0  = mt * 256;
  if (m0 >= cnt) return;                 // uniform: before any barrier
  int base = offs[e] + m0;
  int tid = threadIdx.x;
  int wave = tid >> 5, lane = tid & 31;
  int g = lane >> 4, q = lane & 15;
  int n0 = blockIdx.y * 64;
  int sn = tid >> 2, sp = tid & 3;
  const __bf16* bsrc = w1bf + ((size_t)e * DIM_H + (size_t)(n0 + sn)) * DIM_D + sp * 8;
  int bdoff = sn * 64 + sp * 16;
  int mw = m0 + wave * 32;
  int i0 = (mw + q      < cnt) ? base + wave * 32 + q      : base;
  int i1 = (mw + 16 + q < cnt) ? base + wave * 32 + 16 + q : base;
  const __bf16* a0p = xbf + (size_t)rows[i0] * DIM_D;
  const __bf16* a1p = xbf + (size_t)rows[i1] * DIM_D;

  v8f acc0[4] = {}, acc1[4] = {};
  auto step = [&](int kb, const char* tile) {
    v16bf a0 = load_a_frag(a0p, kb, g);
    v16bf a1 = load_a_frag(a1p, kb, g);
#pragma unroll
    for (int f = 0; f < 4; ++f) {
      v16bf b = load_b_lds(tile, f * 16, q, g);
      acc0[f] = wmma_bf16(a0, b, acc0[f]);
      acc1[f] = wmma_bf16(a1, b, acc1[f]);
    }
  };

  stage16(bsrc, smem + bdoff);
  stage_fence();
  int cur = 0;
  for (int kb = 0; kb < DIM_D - 32; kb += 32) {
    stage16(bsrc + kb + 32, smem + (cur ^ 1) * 4096 + bdoff);
    step(kb, smem + cur * 4096);
    stage_fence();
    cur ^= 1;
  }
  step(DIM_D - 32, smem + cur * 4096);

#pragma unroll
  for (int f = 0; f < 4; ++f)
#pragma unroll
    for (int r = 0; r < 8; ++r) {
      int m = g * 8 + r;
      int col = n0 + f * 16 + q;
      if (mw + m < cnt)
        hbuf[(size_t)(base + wave * 32 + m) * DIM_H + col] = (__bf16)silu_f(acc0[f][r]);
      if (mw + 16 + m < cnt)
        hbuf[(size_t)(base + wave * 32 + 16 + m) * DIM_H + col] = (__bf16)silu_f(acc1[f][r]);
    }
}

// =====================================================================
// Grouped GEMM2: ebuf = h @ w2[e]^T  (f32 per-slot rows)
// =====================================================================
__global__ __launch_bounds__(256) void k_expert_gemm2(
    const __bf16* __restrict__ hbuf, const __bf16* __restrict__ w2bf,
    const int* __restrict__ counts, const int* __restrict__ offs,
    float* __restrict__ ebuf) {
  __shared__ __align__(32) char smem[2 * 4096];
  int e  = blockIdx.x >> 4;
  int mt = blockIdx.x & 15;
  int cnt = counts[e];
  int m0  = mt * 256;
  if (m0 >= cnt) return;
  int base = offs[e] + m0;
  int tid = threadIdx.x;
  int wave = tid >> 5, lane = tid & 31;
  int g = lane >> 4, q = lane & 15;
  int n0 = blockIdx.y * 64;
  int sn = tid >> 2, sp = tid & 3;
  const __bf16* bsrc = w2bf + ((size_t)e * DIM_D + (size_t)(n0 + sn)) * DIM_H + sp * 8;
  int bdoff = sn * 64 + sp * 16;
  int mw = m0 + wave * 32;
  int i0 = (mw + q      < cnt) ? base + wave * 32 + q      : base;
  int i1 = (mw + 16 + q < cnt) ? base + wave * 32 + 16 + q : base;
  const __bf16* a0p = hbuf + (size_t)i0 * DIM_H;
  const __bf16* a1p = hbuf + (size_t)i1 * DIM_H;

  v8f acc0[4] = {}, acc1[4] = {};
  auto step = [&](int kb, const char* tile) {
    v16bf a0 = load_a_frag(a0p, kb, g);
    v16bf a1 = load_a_frag(a1p, kb, g);
#pragma unroll
    for (int f = 0; f < 4; ++f) {
      v16bf b = load_b_lds(tile, f * 16, q, g);
      acc0[f] = wmma_bf16(a0, b, acc0[f]);
      acc1[f] = wmma_bf16(a1, b, acc1[f]);
    }
  };

  stage16(bsrc, smem + bdoff);
  stage_fence();
  int cur = 0;
  for (int kb = 0; kb < DIM_H - 32; kb += 32) {
    stage16(bsrc + kb + 32, smem + (cur ^ 1) * 4096 + bdoff);
    step(kb, smem + cur * 4096);
    stage_fence();
    cur ^= 1;
  }
  step(DIM_H - 32, smem + cur * 4096);

#pragma unroll
  for (int f = 0; f < 4; ++f)
#pragma unroll
    for (int r = 0; r < 8; ++r) {
      int m = g * 8 + r;
      int col = n0 + f * 16 + q;
      if (mw + m < cnt)
        ebuf[(size_t)(base + wave * 32 + m) * DIM_D + col] = acc0[f][r];
      if (mw + 16 + m < cnt)
        ebuf[(size_t)(base + wave * 32 + 16 + m) * DIM_D + col] = acc1[f][r];
    }
}

// =====================================================================
// Shared expert stage 1: g = silu(x@sgGate^T) * (x@sgUp^T) -> bf16
// block tile = 128 M x 64 N; wave tile = 16 M x 64 N (dual accumulators).
// =====================================================================
__global__ __launch_bounds__(256) void k_shared_gemm1(
    const __bf16* __restrict__ xbf, const __bf16* __restrict__ sgGate,
    const __bf16* __restrict__ sgUp, __bf16* __restrict__ gbuf) {
  __shared__ __align__(32) char smem[2 * 8192];   // [buf][gate|up][64n x 64B]
  int mt  = blockIdx.x;
  int tid = threadIdx.x;
  int wave = tid >> 5, lane = tid & 31;
  int g = lane >> 4, q = lane & 15;
  int n0 = blockIdx.y * 64;
  int sn = tid >> 2, sp = tid & 3;
  const __bf16* gsrcG = sgGate + (size_t)(n0 + sn) * DIM_D + sp * 8;
  const __bf16* gsrcU = sgUp   + (size_t)(n0 + sn) * DIM_D + sp * 8;
  int bdoff = sn * 64 + sp * 16;
  const __bf16* aptr = xbf + (size_t)(mt * 128 + wave * 16 + q) * DIM_D;

  v8f accg[4] = {}, accu[4] = {};
  auto step = [&](int kb, const char* tg) {
    const char* tu = tg + 4096;
    v16bf a = load_a_frag(aptr, kb, g);
#pragma unroll
    for (int f = 0; f < 4; ++f) {
      accg[f] = wmma_bf16(a, load_b_lds(tg, f * 16, q, g), accg[f]);
      accu[f] = wmma_bf16(a, load_b_lds(tu, f * 16, q, g), accu[f]);
    }
  };

  stage16(gsrcG, smem + bdoff);
  stage16(gsrcU, smem + 4096 + bdoff);
  stage_fence();
  int cur = 0;
  for (int kb = 0; kb < DIM_D - 32; kb += 32) {
    stage16(gsrcG + kb + 32, smem + (cur ^ 1) * 8192 + bdoff);
    stage16(gsrcU + kb + 32, smem + (cur ^ 1) * 8192 + 4096 + bdoff);
    step(kb, smem + cur * 8192);
    stage_fence();
    cur ^= 1;
  }
  step(DIM_D - 32, smem + cur * 8192);

#pragma unroll
  for (int f = 0; f < 4; ++f)
#pragma unroll
    for (int r = 0; r < 8; ++r) {
      int t = mt * 128 + wave * 16 + g * 8 + r;
      gbuf[(size_t)t * DIM_H + n0 + f * 16 + q] = (__bf16)(silu_f(accg[f][r]) * accu[f][r]);
    }
}

// =====================================================================
// Shared expert stage 2: out = g @ sgDown^T (plain f32 store; runs before combine)
// =====================================================================
__global__ __launch_bounds__(256) void k_shared_gemm2(
    const __bf16* __restrict__ gbuf, const __bf16* __restrict__ sgDown,
    float* __restrict__ out) {
  __shared__ __align__(32) char smem[2 * 4096];
  int mt  = blockIdx.x;
  int tid = threadIdx.x;
  int wave = tid >> 5, lane = tid & 31;
  int g = lane >> 4, q = lane & 15;
  int n0 = blockIdx.y * 64;
  int sn = tid >> 2, sp = tid & 3;
  const __bf16* bsrc = sgDown + (size_t)(n0 + sn) * DIM_H + sp * 8;
  int bdoff = sn * 64 + sp * 16;
  const __bf16* a0p = gbuf + (size_t)(mt * 256 + wave * 32 + q) * DIM_H;
  const __bf16* a1p = gbuf + (size_t)(mt * 256 + wave * 32 + 16 + q) * DIM_H;

  v8f acc0[4] = {}, acc1[4] = {};
  auto step = [&](int kb, const char* tile) {
    v16bf a0 = load_a_frag(a0p, kb, g);
    v16bf a1 = load_a_frag(a1p, kb, g);
#pragma unroll
    for (int f = 0; f < 4; ++f) {
      v16bf b = load_b_lds(tile, f * 16, q, g);
      acc0[f] = wmma_bf16(a0, b, acc0[f]);
      acc1[f] = wmma_bf16(a1, b, acc1[f]);
    }
  };

  stage16(bsrc, smem + bdoff);
  stage_fence();
  int cur = 0;
  for (int kb = 0; kb < DIM_H - 32; kb += 32) {
    stage16(bsrc + kb + 32, smem + (cur ^ 1) * 4096 + bdoff);
    step(kb, smem + cur * 4096);
    stage_fence();
    cur ^= 1;
  }
  step(DIM_H - 32, smem + cur * 4096);

#pragma unroll
  for (int f = 0; f < 4; ++f)
#pragma unroll
    for (int r = 0; r < 8; ++r) {
      int m = g * 8 + r;
      int col = n0 + f * 16 + q;
      out[(size_t)(mt * 256 + wave * 32 + m) * DIM_D + col]      = acc0[f][r];
      out[(size_t)(mt * 256 + wave * 32 + 16 + m) * DIM_D + col] = acc1[f][r];
    }
}

// ---------------- Deterministic combine (no f32 atomics) ----------------
__global__ __launch_bounds__(256) void k_combine(
    const float* __restrict__ ebuf, const int* __restrict__ slot0,
    const int* __restrict__ slot1, const float* __restrict__ w0t,
    const float* __restrict__ w1t, float* __restrict__ out) {
  int i = blockIdx.x * 256 + threadIdx.x;
  int t = i >> 9;             // / DIM_D (512)
  int d = i & (DIM_D - 1);
  float r = out[i];           // shared-expert value written by k_shared_gemm2
  r += w0t[t] * ebuf[(size_t)slot0[t] * DIM_D + d];
  r += w1t[t] * ebuf[(size_t)slot1[t] * DIM_D + d];
  out[i] = r;
}

extern "C" void kernel_launch(void* const* d_in, const int* in_sizes, int n_in,
                              void* d_out, int out_size, void* d_ws, size_t ws_size,
                              hipStream_t stream) {
  (void)in_sizes; (void)n_in; (void)out_size; (void)ws_size;
  const float* x      = (const float*)d_in[0];
  const float* gate_w = (const float*)d_in[1];
  const float* w1     = (const float*)d_in[2];
  const float* w2     = (const float*)d_in[3];
  const float* sgg    = (const float*)d_in[4];
  const float* sgu    = (const float*)d_in[5];
  const float* sgd    = (const float*)d_in[6];
  float* out = (float*)d_out;

  char* ws = (char*)d_ws;
  size_t off = 0;
  auto carve = [&](size_t bytes) -> void* {
    void* p = ws + off;
    off += (bytes + 255) & ~(size_t)255;
    return p;
  };
  __bf16* xbf  = (__bf16*)carve((size_t)TOK_N * DIM_D * 2);
  __bf16* w1bf = (__bf16*)carve((size_t)NUM_E * DIM_H * DIM_D * 2);
  __bf16* w2bf = (__bf16*)carve((size_t)NUM_E * DIM_D * DIM_H * 2);
  __bf16* sggb = (__bf16*)carve((size_t)DIM_H * DIM_D * 2);
  __bf16* sgub = (__bf16*)carve((size_t)DIM_H * DIM_D * 2);
  __bf16* sgdb = (__bf16*)carve((size_t)DIM_D * DIM_H * 2);
  __bf16* hbuf = (__bf16*)carve((size_t)NK_TOTAL * DIM_H * 2);
  __bf16* gbuf = (__bf16*)carve((size_t)TOK_N * DIM_H * 2);
  float*  ebuf = (float*) carve((size_t)NK_TOTAL * DIM_D * 4);
  int*  rows  = (int*)  carve(NK_TOTAL * 4);
  int*  idx0  = (int*)  carve(TOK_N * 4);
  int*  idx1  = (int*)  carve(TOK_N * 4);
  float* w0t  = (float*)carve(TOK_N * 4);
  float* w1t  = (float*)carve(TOK_N * 4);
  int*  slot0 = (int*)  carve(TOK_N * 4);
  int*  slot1 = (int*)  carve(TOK_N * 4);
  int*  counts= (int*)  carve(NUM_E * 4);
  int*  offs  = (int*)  carve(NUM_E * 4);
  int*  fill  = (int*)  carve(NUM_E * 4);

  auto cvt = [&](const float* s, __bf16* d, int n) {
    k_cvt_bf16<<<(n + 255) / 256, 256, 0, stream>>>(s, d, n);
  };
  cvt(x,   xbf,  TOK_N * DIM_D);
  cvt(w1,  w1bf, NUM_E * DIM_H * DIM_D);
  cvt(w2,  w2bf, NUM_E * DIM_D * DIM_H);
  cvt(sgg, sggb, DIM_H * DIM_D);
  cvt(sgu, sgub, DIM_H * DIM_D);
  cvt(sgd, sgdb, DIM_D * DIM_H);

  k_zero_i32<<<1, 32, 0, stream>>>(counts, NUM_E);
  k_zero_i32<<<1, 32, 0, stream>>>(fill,   NUM_E);

  k_router<<<TOK_N * 32 / 256, 256, 0, stream>>>(x, gate_w, idx0, idx1, w0t, w1t, counts);
  k_offsets<<<1, 1, 0, stream>>>(counts, offs);
  k_fill<<<TOK_N / 256, 256, 0, stream>>>(idx0, idx1, fill, offs, rows, slot0, slot1);

  k_expert_gemm1<<<dim3(NUM_E * 16, DIM_H / 64), 256, 0, stream>>>(
      xbf, w1bf, rows, counts, offs, hbuf);
  k_expert_gemm2<<<dim3(NUM_E * 16, DIM_D / 64), 256, 0, stream>>>(
      hbuf, w2bf, counts, offs, ebuf);

  k_shared_gemm1<<<dim3(TOK_N / 128, DIM_H / 64), 256, 0, stream>>>(xbf, sggb, sgub, gbuf);
  k_shared_gemm2<<<dim3(TOK_N / 256, DIM_D / 64), 256, 0, stream>>>(gbuf, sgdb, out);

  k_combine<<<TOK_N * DIM_D / 256, 256, 0, stream>>>(ebuf, slot0, slot1, w0t, w1t, out);
}
